// GraphSAGELSTM_18167711662123
// MI455X (gfx1250) — matrix operand order repaired and checked by reading, not compiled, not run on hardware
//
#include <hip/hip_runtime.h>
#include <hip/hip_bf16.h>
#include <stdint.h>

typedef __attribute__((ext_vector_type(16))) _Float16     v16h;
typedef __attribute__((ext_vector_type(8)))  _Float16     v8h;
typedef __attribute__((ext_vector_type(8)))  float        v8f;
typedef __attribute__((ext_vector_type(4)))  unsigned int v4u;
typedef __attribute__((ext_vector_type(8)))  int          v8i;
typedef __attribute__((ext_vector_type(4)))  int          v4i;

#define NN 2048
#define DD 64
#define BB 8
#define TI 128   // i-tile (output rows) per workgroup
#define TK 32    // k-chunk (j dimension) per WMMA step

__device__ __forceinline__ float sigmoidf_(float v) { return 1.0f / (1.0f + expf(-v)); }

// ---------------------------------------------------------------------------
// Tensor Data Mover: 2-D tile global -> LDS (CDNA5 ISA)
// D# group0: count=1 | lds_addr | global_addr[56:0] | type=2
// D# group1: data_size=4B, tensor_dim0/1, tile_dim0/1, tensor_dim0_stride
// This toolchain exposes the 6-arg builtin: (v4u, v8i, v4i, v4i, v8i, i32 cpol)
// ---------------------------------------------------------------------------
#if defined(__has_builtin)
#  if __has_builtin(__builtin_amdgcn_tensor_load_to_lds)
#    define HAVE_TDM 1
#  endif
#endif
#ifndef HAVE_TDM
#  define HAVE_TDM 0
#endif

#if HAVE_TDM
__device__ __forceinline__ void tdm_load_2d(unsigned lds_off, const float* gptr,
                                            unsigned tile_d0, unsigned tile_d1,
                                            unsigned stride0,
                                            unsigned tensor_d0, unsigned tensor_d1) {
  unsigned long long ga = (unsigned long long)(uintptr_t)gptr;
  v4u g0;
  g0[0] = 1u;                                              // count = 1 (valid)
  g0[1] = lds_off;                                         // LDS byte address
  g0[2] = (unsigned)(ga & 0xFFFFFFFFull);                  // global addr lo
  g0[3] = (unsigned)((ga >> 32) & 0x01FFFFFFull) | 0x80000000u; // hi + type=2
  v8i g1;
  g1[0] = (int)(2u << 16);                                 // data_size = 4 bytes
  g1[1] = (int)((tensor_d0 & 0xFFFFu) << 16);              // tensor_dim0 lo16
  g1[2] = (int)(((tensor_d0 >> 16) & 0xFFFFu) | ((tensor_d1 & 0xFFFFu) << 16));
  g1[3] = (int)(((tensor_d1 >> 16) & 0xFFFFu) | ((tile_d0 & 0xFFFFu) << 16));
  g1[4] = (int)(tile_d1 & 0xFFFFu);                        // tile_dim1, tile_dim2=0
  g1[5] = (int)stride0;                                    // tensor_dim0_stride lo32
  g1[6] = 0;                                               // stride hi / dim1_stride
  g1[7] = 0;
  v4i z4 = {0, 0, 0, 0};
  v8i z8 = {0, 0, 0, 0, 0, 0, 0, 0};
  __builtin_amdgcn_tensor_load_to_lds(g0, g1, z4, z4, z8, 0);
}
#endif

// ---------------------------------------------------------------------------
// Kernel A: per (b, r in {0,1}) -- argmin gather + tiny LSTM step -> hbuf
// ---------------------------------------------------------------------------
__global__ __launch_bounds__(256) void lstm_prep(
    const float* __restrict__ x, const float* __restrict__ adj,
    const float* __restrict__ order_adj, const float* __restrict__ w_ih,
    const float* __restrict__ b_ih, const float* __restrict__ b_hh,
    float* __restrict__ hbuf) {
  __shared__ float skey[256];
  __shared__ int   sidx[256];
  __shared__ float wx[DD];
  __shared__ float gates[256];
  const int b = blockIdx.x >> 1;
  const int r = blockIdx.x & 1;
  const int i = NN - 2 + r;
  const int t = threadIdx.x;

  float bk = 1e30f;
  int   bj = 0;
  for (int j = t; j < NN; j += 256) {
    float mv = adj[((size_t)b * NN + j) * NN + i];
    if (mv == 1.0f) {
      float key = order_adj[(size_t)j * NN + i];
      if (key < bk || (key == bk && j < bj)) { bk = key; bj = j; }
    }
  }
  skey[t] = bk; sidx[t] = bj;
  __syncthreads();
  for (int s = 128; s > 0; s >>= 1) {
    if (t < s) {
      float ok = skey[t + s]; int oj = sidx[t + s];
      if (ok < skey[t] || (ok == skey[t] && oj < sidx[t])) { skey[t] = ok; sidx[t] = oj; }
    }
    __syncthreads();
  }
  const int  jmin = sidx[0];
  const bool has  = skey[0] < 1e30f;
  const float wsel = adj[((size_t)b * NN + jmin) * NN + i];
  if (t < DD) wx[t] = wsel * x[((size_t)b * NN + jmin) * DD + t];
  __syncthreads();
  float g = b_ih[t] + b_hh[t];
  #pragma unroll 8
  for (int d = 0; d < DD; ++d) g += wx[d] * w_ih[t * DD + d];
  gates[t] = g;
  __syncthreads();
  if (t < DD) {
    float ig = gates[t];
    float gg = gates[128 + t];
    float og = gates[192 + t];
    float cc = sigmoidf_(ig) * tanhf(gg);
    float hh = sigmoidf_(og) * tanhf(cc);
    hbuf[(b * 2 + r) * DD + t] = has ? hh : 0.0f;
  }
}

// ---------------------------------------------------------------------------
// Kernel B: WMMA main GEMM  sum_x = new_adj^T @ x, fused with mean/lstm rows,
// output projections (@weight, @bias) and row L2-normalization.
// Grid: BB * (NN/TI) blocks, 256 threads (8 waves); wave w owns rows w*16..+15.
//
// LDS map (bytes, 64 KB total):
//   main loop:  [0..16K)  adjT f32 32x128   [16K..32K) wadjT f32 32x128
//               [32K..40K) xT f32 32x64
//               [40K..48K) adjH f16 [i][k] 128x32   [48K..52K) xH f16 [d][k] 64x32
//   epilogue:   [0..32K) sx f32 128x64   [32K..48K) wgt   [48K..64K) bsd
// ---------------------------------------------------------------------------
__global__ __launch_bounds__(256) void fused_graph_kernel(
    const float* __restrict__ x, const float* __restrict__ adj,
    const float* __restrict__ inv_deg, const float* __restrict__ weight,
    const float* __restrict__ bias, const float* __restrict__ weight_adj,
    const float* __restrict__ hbuf, float* __restrict__ out) {
  __shared__ float smem[16384];          // 64 KB
  float*    adjT  = smem;                // 32 x 128 f32 (main loop)
  float*    wadjT = smem + 4096;         // 32 x 128 f32
  float*    xT    = smem + 8192;         // 32 x  64 f32
  _Float16* adjH  = (_Float16*)(smem + 10240); // 128 x 32 f16 (K contiguous)
  _Float16* xH    = (_Float16*)(smem + 12288); //  64 x 32 f16 (K contiguous)
  float*    sx    = smem;                // 128 x 64 f32 (epilogue)
  float*    wgt   = smem + 8192;
  float*    bsd   = smem + 12288;

  const int tid  = threadIdx.x;
  const int lane = tid & 31;
  const int wv   = tid >> 5;
  const int b    = blockIdx.x >> 4;
  const int i0   = (blockIdx.x & 15) * TI;

  const float* adjB = adj + (size_t)b * NN * NN;
  const float* xB   = x   + (size_t)b * NN * DD;

  v8f acc[4];
  #pragma unroll
  for (int t = 0; t < 4; ++t)
    #pragma unroll
    for (int e = 0; e < 8; ++e) acc[t][e] = 0.0f;

  // WMMA lane decomposition (16-bit A 16x32 / B 32x16 layouts, CDNA5 ISA)
  const int mA   = lane & 15;
  const int selA = lane >> 4;            // K-half select for A
  const int iLoc = wv * 16 + mA;         // local output row this lane feeds
  const int nB   = lane & 15;
  const int kbB  = (lane >> 4) * 16;     // K-half base for B

  const unsigned lds_adj  = (unsigned)(uintptr_t)(void*)adjT;
  const unsigned lds_wadj = (unsigned)(uintptr_t)(void*)wadjT;
  const unsigned lds_x    = (unsigned)(uintptr_t)(void*)xT;

  for (int j0 = 0; j0 < NN; j0 += TK) {
#if HAVE_TDM
    if (tid == 0) {
      tdm_load_2d(lds_adj,  adjB       + (size_t)j0 * NN + i0, TI, TK, NN, NN, NN);
      tdm_load_2d(lds_wadj, weight_adj + (size_t)j0 * NN + i0, TI, TK, NN, NN, NN);
      tdm_load_2d(lds_x,    xB         + (size_t)j0 * DD,      DD, TK, DD, DD, NN);
      __builtin_amdgcn_s_wait_tensorcnt(0);
    }
#else
    for (int idx = tid; idx < TK * TI; idx += 256) {
      int kk = idx >> 7, ii = idx & 127;
      adjT[idx]  = adjB[(size_t)(j0 + kk) * NN + i0 + ii];
      wadjT[idx] = weight_adj[(size_t)(j0 + kk) * NN + i0 + ii];
    }
    for (int idx = tid; idx < TK * DD; idx += 256)
      xT[idx] = xB[(size_t)j0 * DD + idx];
#endif
    __syncthreads();   // tiles ready; previous iteration's f16 reads complete

    if (j0 + TK < NN)
      __builtin_prefetch(adjB + (size_t)(j0 + TK) * NN + i0 + lane * 4, 0, 1);

    // ---- shared conversion pass: f32 tiles -> transposed f16 staging ----
    // new_adj with masked weighting: interior (i<N-2 && j<N-2) gets *weight_adj
    #pragma unroll
    for (int s = 0; s < (TK * TI) / 256; ++s) {
      int idx = tid + s * 256;
      int kk = idx >> 7, ii = idx & 127;
      float a = adjT[idx], w = wadjT[idx];
      bool interior = ((i0 + ii) < NN - 2) && ((j0 + kk) < NN - 2);
      adjH[ii * TK + kk] = (_Float16)(interior ? a * w : a);
    }
    #pragma unroll
    for (int s = 0; s < (TK * DD) / 256; ++s) {
      int idx = tid + s * 256;
      int kk = idx >> 6, cc = idx & 63;
      xH[cc * TK + kk] = (_Float16)xT[idx];
    }
    __syncthreads();   // f16 staging ready; f32 tiles free for next TDM

    // ---- fragments: wide LDS loads, K contiguous ----
    v16h af;
    {
      const _Float16* ap = adjH + iLoc * TK;
      v8h alo = *(const v8h*)(ap + selA * 16);       // K = selA*16 .. +7
      v8h ahi = *(const v8h*)(ap + selA * 16 + 8);   // K = selA*16+8 .. +7
      #pragma unroll
      for (int e = 0; e < 8; ++e) { af[e] = alo[e]; af[8 + e] = ahi[e]; }
    }
    #pragma unroll
    for (int t = 0; t < 4; ++t) {
      v16h bf = *(const v16h*)(xH + (t * 16 + nB) * TK + kbB);
      acc[t] = __builtin_amdgcn_wmma_f32_16x16x32_f16(false, af, false, bf,
                                                      (short)0, acc[t],
                                                      false, false);
    }
  }
  __syncthreads();

  // --- store C tiles to LDS with inv_deg scaling / LSTM row substitution ---
  {
    const int nn = lane & 15;
    const int mh = (lane >> 4) * 8;      // C layout: lanes 16-31 hold M=r+8
    #pragma unroll
    for (int r = 0; r < 8; ++r) {
      int row = wv * 16 + mh + r;
      int gi  = i0 + row;
      bool isl = gi >= NN - 2;
      float inv = isl ? 0.0f
                      : inv_deg[(size_t)b * NN * NN + (size_t)gi * NN + gi];
      #pragma unroll
      for (int t = 0; t < 4; ++t) {
        int col = t * 16 + nn;
        float vV = acc[t][r];
        vV = isl ? hbuf[(b * 2 + (gi - (NN - 2))) * DD + col] : vV * inv;
        sx[row * DD + col] = vV;
      }
    }
  }
  for (int idx = tid; idx < DD * DD; idx += 256) {
    wgt[idx] = weight[idx];
    bsd[idx] = bias[idx];
  }
  __syncthreads();

  // --- per-row projections + L2 normalization (2 threads per row) ---
  const int row  = tid >> 1;
  const int coff = (tid & 1) * 32;
  const int gi   = i0 + row;
  const float* srow = sx + row * DD;
  const float* xrow = xB + (size_t)gi * DD;
  float accN[32], accS[32];
  #pragma unroll
  for (int c = 0; c < 32; ++c) { accN[c] = 0.0f; accS[c] = 0.0f; }
  for (int k = 0; k < DD; ++k) {
    float a  = srow[k];
    float xv = xrow[k];
    const float4* wr = (const float4*)(wgt + k * DD + coff);
    const float4* br = (const float4*)(bsd + k * DD + coff);
    #pragma unroll
    for (int q = 0; q < 8; ++q) {
      float4 wq = wr[q], bq = br[q];
      accN[4 * q + 0] += a * wq.x;  accN[4 * q + 1] += a * wq.y;
      accN[4 * q + 2] += a * wq.z;  accN[4 * q + 3] += a * wq.w;
      accS[4 * q + 0] += xv * bq.x; accS[4 * q + 1] += xv * bq.y;
      accS[4 * q + 2] += xv * bq.z; accS[4 * q + 3] += xv * bq.w;
    }
  }
  float ssq = 0.0f;
  #pragma unroll
  for (int c = 0; c < 32; ++c) ssq += accN[c] * accN[c] + accS[c] * accS[c];
  float tot = ssq + __shfl_xor(ssq, 1);          // partner thread of same row
  float s = 1.0f / fmaxf(sqrtf(tot), 1e-12f);
  float* orow = out + ((size_t)b * NN + gi) * 128;
  #pragma unroll
  for (int c = 0; c < 32; ++c) {
    orow[coff + c]      = accN[c] * s;
    orow[64 + coff + c] = accS[c] * s;
  }
}

// ---------------------------------------------------------------------------
extern "C" void kernel_launch(void* const* d_in, const int* in_sizes, int n_in,
                              void* d_out, int out_size, void* d_ws, size_t ws_size,
                              hipStream_t stream) {
  (void)in_sizes; (void)n_in; (void)out_size; (void)ws_size;
  const float* x          = (const float*)d_in[0];
  const float* adj        = (const float*)d_in[1];
  const float* inv_deg    = (const float*)d_in[2];
  const float* weight     = (const float*)d_in[3];
  const float* bias       = (const float*)d_in[4];
  const float* weight_adj = (const float*)d_in[5];
  const float* order_adj  = (const float*)d_in[6];
  const float* w_ih       = (const float*)d_in[7];
  // d_in[8] = w_hh: unused by the reference math (no hidden state)
  const float* b_ih       = (const float*)d_in[9];
  const float* b_hh       = (const float*)d_in[10];
  float* hbuf = (float*)d_ws;  // 8*2*64 floats

  lstm_prep<<<BB * 2, 256, 0, stream>>>(x, adj, order_adj, w_ih, b_ih, b_hh, hbuf);
  fused_graph_kernel<<<BB * (NN / TI), 256, 0, stream>>>(
      x, adj, inv_deg, weight, bias, weight_adj, hbuf, (float*)d_out);
}
